// RecurrentGCN_37391985279605
// MI455X (gfx1250) — matrix operand order repaired and checked by reading, not compiled, not run on hardware
//
#include <hip/hip_runtime.h>
#include <hip/hip_bf16.h>

typedef __attribute__((ext_vector_type(16))) _Float16 v16h;
typedef __attribute__((ext_vector_type(8)))  _Float16 v8h;
typedef __attribute__((ext_vector_type(2)))  _Float16 v2h;
typedef __attribute__((ext_vector_type(8)))  float    v8f;
typedef __attribute__((ext_vector_type(4)))  float    v4f;

#define FIN   10
#define HID   32
#define DIN   42   // FIN + HID
#define KPAD  64   // padded K (two 32-wide WMMA k-steps)
#define WPB   8    // waves per 256-thread block

// ---------------------------------------------------------------------------
// Prep: wbuf[g][n][k] = f16(W_g[0][k][n] + W_g[1][k][n]), k-contiguous (64,
// zero padded for k>=42). Column-major so a lane's B fragment (16 contiguous
// k for one column n) is a single contiguous 32B read.
// ---------------------------------------------------------------------------
__global__ void dcrnn_prep_weights(const float* __restrict__ Wz,
                                   const float* __restrict__ Wr,
                                   const float* __restrict__ Wh,
                                   _Float16* __restrict__ wbuf) {
    int idx = blockIdx.x * blockDim.x + threadIdx.x;
    const int total = 3 * HID * KPAD;
    for (int i = idx; i < total; i += gridDim.x * blockDim.x) {
        int g = i >> 11;            // / (32*64)
        int n = (i >> 6) & 31;
        int k = i & 63;
        const float* W = (g == 0) ? Wz : ((g == 1) ? Wr : Wh);
        float v = 0.0f;
        if (k < DIN) v = W[k * HID + n] + W[DIN * HID + k * HID + n];
        wbuf[i] = (_Float16)v;
    }
}

// Fast transcendentals: single v_exp_f32 + v_rcp_f32, no IEEE div refinement.
__device__ __forceinline__ float sigmoid_f(float x) {
    return __builtin_amdgcn_rcpf(1.0f + __expf(-x));
}
__device__ __forceinline__ float tanh_f(float x) {
    // tanh(x) = 1 - 2/(exp(2x)+1)
    return __builtin_fmaf(-2.0f, __builtin_amdgcn_rcpf(__expf(2.0f * x) + 1.0f), 1.0f);
}

__device__ __forceinline__ v16h wmma_cat(v8h g0, v8h g1) {
    v16h a;
#pragma unroll
    for (int e = 0; e < 8; ++e) { a[e] = g0[e]; a[8 + e] = g1[e]; }
    return a;
}

// ---------------------------------------------------------------------------
// Main kernel: one wave per 16-node tile via f16 WMMA. FULL TILES ONLY —
// no clamps, no store guards, so all global accesses fold into single-base
// + immediate-offset addressing.
// ---------------------------------------------------------------------------
__global__ __launch_bounds__(256) void dcrnn_gru_kernel(
    const float* __restrict__ x, const float* __restrict__ h,
    const float* __restrict__ bz, const float* __restrict__ br,
    const float* __restrict__ bh, const float* __restrict__ wlin,
    const float* __restrict__ blin, const _Float16* __restrict__ wbuf,
    float* __restrict__ out_head, float* __restrict__ out_h, int numTiles)
{
    __shared__ __align__(16) _Float16 wlds[3 * HID * KPAD];   // 12KB: fused f16 weights
    __shared__ __align__(16) float    cb[32 * 3 + 64 + 2];    // bz|br|bh|wlin|blin
    __shared__ __align__(16) _Float16 xh[WPB][16 * KPAD];     // 16KB: per-wave [x | r*h | pad]
    __shared__ __align__(16) float    hnew[WPB][16 * HID];    // 16KB: per-wave h_new (f32)

    const int tid = threadIdx.x;
    for (int i = tid; i < 3 * HID * KPAD; i += 256) wlds[i] = wbuf[i];
    if (tid < 32) { cb[tid] = bz[tid]; cb[32 + tid] = br[tid]; cb[64 + tid] = bh[tid]; }
    if (tid < 64) cb[96 + tid] = wlin[tid];
    if (tid < 2)  cb[160 + tid] = blin[tid];
    __syncthreads();

    const int lane = tid & 31;
    const int wv   = tid >> 5;
    const int hl   = lane >> 4;    // half-wave (0/1)
    const int mrow = lane & 15;    // M row / N column within tile

    _Float16* myxh = xh[wv];
    float*    myh  = hnew[wv];

    // zero the K padding (cols 42..63) once; never rewritten inside the loop
    for (int c = DIN + hl; c < KPAD; c += 2) myxh[mrow * KPAD + c] = (_Float16)0.0f;

    // B fragments resident in registers across all tiles:
    // lane holds n = 16*t + mrow, k = s*32 + hl*16 + [0..15] (contiguous in wlds)
    v16h Bz[2][2], Br[2][2], Bh[2][2];
#pragma unroll
    for (int t = 0; t < 2; ++t)
#pragma unroll
        for (int s = 0; s < 2; ++s) {
            int off = (16 * t + mrow) * KPAD + s * 32 + hl * 16;
            Bz[t][s] = *(const v16h*)(wlds + 0 * 2048 + off);
            Br[t][s] = *(const v16h*)(wlds + 1 * 2048 + off);
            Bh[t][s] = *(const v16h*)(wlds + 2 * 2048 + off);
        }

    // tile-invariant per-lane biases (C-layout column n = t*16 + mrow)
    float bzn[2], brn[2], bhn[2];
#pragma unroll
    for (int t = 0; t < 2; ++t) {
        int n = t * 16 + mrow;
        bzn[t] = cb[n]; brn[t] = cb[32 + n]; bhn[t] = cb[64 + n];
    }
    const float blin_l = cb[160 + hl];

    const int wavesTotal = gridDim.x * WPB;
    const int waveId     = blockIdx.x * WPB + wv;

    for (int tile = waveId; tile < numTiles; tile += wavesTotal) {
        const int base = tile << 4;

        { // prefetch next tile's h row for this lane
            int nt = tile + wavesTotal;
            if (nt < numTiles)
                __builtin_prefetch(h + (size_t)((nt << 4) + mrow) * HID, 0, 0);
        }

        // ---- h in C-layout straight from global (coalesced: lane = column n);
        // one base pointer, 16 immediate-offset b32 loads in a clause.
        float hC[2][8];
        {
            const float* hp = h + (size_t)(base + hl * 8) * HID + mrow;
#pragma unroll
            for (int t = 0; t < 2; ++t)
#pragma unroll
                for (int j = 0; j < 8; ++j)
                    hC[t][j] = hp[j * HID + t * 16];
        }

        // ---- stage tile: h -> xh cols 10..41 (f16); x -> cols 0..9
        {
            const int node = base + mrow;
            const float* hp = h + (size_t)node * HID + hl * 16;
#pragma unroll
            for (int q = 0; q < 4; ++q) {
                v4f hv = *(const v4f*)(hp + q * 4);
                int c0 = FIN + hl * 16 + q * 4;
                v2h p0; p0.x = (_Float16)hv.x; p0.y = (_Float16)hv.y;
                v2h p1; p1.x = (_Float16)hv.z; p1.y = (_Float16)hv.w;
                *(v2h*)(myxh + mrow * KPAD + c0)     = p0;
                *(v2h*)(myxh + mrow * KPAD + c0 + 2) = p1;
            }
            const float* xp = x + (size_t)node * FIN + hl * 5;
#pragma unroll
            for (int q = 0; q < 5; ++q)
                myxh[mrow * KPAD + hl * 5 + q] = (_Float16)xp[q];
        }

        // ---- A fragments for xh (contiguous k order: lane half hl holds k = 16*hl + [0..15] per k-step)
        v16h A0, A1;
        {
            const _Float16* row = myxh + mrow * KPAD;
            A0 = wmma_cat(*(const v8h*)(row + hl * 16),      *(const v8h*)(row + 8 + hl * 16));
            A1 = wmma_cat(*(const v8h*)(row + 32 + hl * 16), *(const v8h*)(row + 40 + hl * 16));
        }

        // ---- z and r gates: 8 WMMAs, fp32 accumulate
        v8f accz[2], accr[2];
#pragma unroll
        for (int t = 0; t < 2; ++t) {
            v8f c0 = {};
            c0 = __builtin_amdgcn_wmma_f32_16x16x32_f16(false, A0, false, Bz[t][0], (short)0, c0, false, false);
            c0 = __builtin_amdgcn_wmma_f32_16x16x32_f16(false, A1, false, Bz[t][1], (short)0, c0, false, false);
            accz[t] = c0;
            v8f c1 = {};
            c1 = __builtin_amdgcn_wmma_f32_16x16x32_f16(false, A0, false, Br[t][0], (short)0, c1, false, false);
            c1 = __builtin_amdgcn_wmma_f32_16x16x32_f16(false, A1, false, Br[t][1], (short)0, c1, false, false);
            accr[t] = c1;
        }

        // ---- r*h back into xh's h-region (C layout: m = j + 8*hl, n = 16t+mrow)
#pragma unroll
        for (int t = 0; t < 2; ++t) {
            int n = t * 16 + mrow;
#pragma unroll
            for (int j = 0; j < 8; ++j) {
                int m = j + hl * 8;
                float r  = sigmoid_f(accr[t][j] + brn[t]);
                float rh = r * hC[t][j];
                myxh[m * KPAD + FIN + n] = (_Float16)rh;
            }
        }

        // ---- rebuild A with [x | r*h] and compute h_tilde
        {
            const _Float16* row = myxh + mrow * KPAD;
            A0 = wmma_cat(*(const v8h*)(row + hl * 16),      *(const v8h*)(row + 8 + hl * 16));
            A1 = wmma_cat(*(const v8h*)(row + 32 + hl * 16), *(const v8h*)(row + 40 + hl * 16));
        }
        v8f acch[2];
#pragma unroll
        for (int t = 0; t < 2; ++t) {
            v8f c0 = {};
            c0 = __builtin_amdgcn_wmma_f32_16x16x32_f16(false, A0, false, Bh[t][0], (short)0, c0, false, false);
            c0 = __builtin_amdgcn_wmma_f32_16x16x32_f16(false, A1, false, Bh[t][1], (short)0, c0, false, false);
            acch[t] = c0;
        }

        // ---- gating: h_new into LDS (transpose stage), still in registers/C-layout
#pragma unroll
        for (int t = 0; t < 2; ++t) {
            int n = t * 16 + mrow;
#pragma unroll
            for (int j = 0; j < 8; ++j) {
                int m = j + hl * 8;
                float zv = sigmoid_f(accz[t][j] + bzn[t]);
                float ht = tanh_f(acch[t][j] + bhn[t]);
                float hn = zv * hC[t][j] + (1.0f - zv) * ht;
                myh[m * HID + n] = hn;
            }
        }

        // ---- row-major pass: coalesced b128 h_new stores + fused linear head.
        // lane covers (node = base+mrow, cols hl*16..hl*16+15); head output o = hl.
        {
            const int node = base + mrow;
            const float* hr  = myh + mrow * HID + hl * 16;
            const float* wl0 = cb + 96 + hl * 16;          // wlin[0][cols]
            const float* wl1 = cb + 96 + 32 + hl * 16;     // wlin[1][cols]
            float* op = out_h + (size_t)node * HID + hl * 16;
            float p0 = 0.0f, p1 = 0.0f;
#pragma unroll
            for (int q = 0; q < 4; ++q) {
                v4f hv = *(const v4f*)(hr + q * 4);
                *(v4f*)(op + q * 4) = hv;
                v4f rv; // ReLU
                rv.x = hv.x > 0.0f ? hv.x : 0.0f;
                rv.y = hv.y > 0.0f ? hv.y : 0.0f;
                rv.z = hv.z > 0.0f ? hv.z : 0.0f;
                rv.w = hv.w > 0.0f ? hv.w : 0.0f;
                v4f w0 = *(const v4f*)(wl0 + q * 4);
                v4f w1 = *(const v4f*)(wl1 + q * 4);
                p0 += rv.x * w0.x + rv.y * w0.y + rv.z * w0.z + rv.w * w0.w;
                p1 += rv.x * w1.x + rv.y * w1.y + rv.z * w1.z + rv.w * w1.w;
            }
            // combine the two 16-wide halves across half-waves
            float t0 = p0 + __shfl_xor(p0, 16, 32);
            float t1 = p1 + __shfl_xor(p1, 16, 32);
            float s  = blin_l + (hl ? t1 : t0);
            out_head[(size_t)node * 2 + hl] = s;
        }
    }
}

// ---------------------------------------------------------------------------
// Scalar tail: handles nNodes % 16 leftover nodes (never launched for N=500k).
// One thread per node; plain VALU math, negligible work (< 16 nodes).
// ---------------------------------------------------------------------------
__global__ void dcrnn_tail_kernel(
    const float* __restrict__ x, const float* __restrict__ h,
    const float* __restrict__ Wz, const float* __restrict__ bz,
    const float* __restrict__ Wr, const float* __restrict__ br,
    const float* __restrict__ Wh, const float* __restrict__ bh,
    const float* __restrict__ wlin, const float* __restrict__ blin,
    float* __restrict__ out_head, float* __restrict__ out_h,
    int startNode, int nNodes)
{
    int node = startNode + blockIdx.x * blockDim.x + threadIdx.x;
    if (node >= nNodes) return;
    float xhv[DIN];
    for (int k = 0; k < FIN; ++k) xhv[k] = x[(size_t)node * FIN + k];
    for (int k = 0; k < HID; ++k) xhv[FIN + k] = h[(size_t)node * HID + k];
    float zv[HID], rv[HID];
    for (int o = 0; o < HID; ++o) {
        float az = bz[o], ar = br[o];
        for (int k = 0; k < DIN; ++k) {
            float wz = Wz[k * HID + o] + Wz[DIN * HID + k * HID + o];
            float wr = Wr[k * HID + o] + Wr[DIN * HID + k * HID + o];
            az += xhv[k] * wz; ar += xhv[k] * wr;
        }
        zv[o] = 1.0f / (1.0f + __expf(-az));
        rv[o] = 1.0f / (1.0f + __expf(-ar));
    }
    float xrh[DIN];
    for (int k = 0; k < FIN; ++k) xrh[k] = xhv[k];
    for (int k = 0; k < HID; ++k) xrh[FIN + k] = rv[k] * xhv[FIN + k];
    float hnv[HID];
    for (int o = 0; o < HID; ++o) {
        float a = bh[o];
        for (int k = 0; k < DIN; ++k) {
            float w = Wh[k * HID + o] + Wh[DIN * HID + k * HID + o];
            a += xrh[k] * w;
        }
        float e2 = __expf(2.0f * a);
        float th = (e2 - 1.0f) / (e2 + 1.0f);
        float hn = zv[o] * xhv[FIN + o] + (1.0f - zv[o]) * th;
        hnv[o] = hn;
        out_h[(size_t)node * HID + o] = hn;
    }
    for (int o = 0; o < 2; ++o) {
        float s = blin[o];
        for (int k = 0; k < HID; ++k) {
            float rl = hnv[k] > 0.0f ? hnv[k] : 0.0f;
            s += rl * wlin[o * HID + k];
        }
        out_head[(size_t)node * 2 + o] = s;
    }
}

extern "C" void kernel_launch(void* const* d_in, const int* in_sizes, int n_in,
                              void* d_out, int out_size, void* d_ws, size_t ws_size,
                              hipStream_t stream) {
    const float* x    = (const float*)d_in[0];
    // d_in[1] edge_index (int64) and d_in[2] edge_weight are unused (K=1 DConv)
    const float* h    = (const float*)d_in[3];
    const float* Wz   = (const float*)d_in[4];
    const float* bz   = (const float*)d_in[5];
    const float* Wr   = (const float*)d_in[6];
    const float* br   = (const float*)d_in[7];
    const float* Wh   = (const float*)d_in[8];
    const float* bh   = (const float*)d_in[9];
    const float* wlin = (const float*)d_in[10];
    const float* blin = (const float*)d_in[11];

    const int nNodes = in_sizes[0] / FIN;            // 500000
    float* out_head = (float*)d_out;                 // [N,2] flat, first
    float* out_h    = out_head + (size_t)nNodes * 2; // then [N,32] flat

    _Float16* wbuf = (_Float16*)d_ws;                // 3*32*64 f16 = 12KB

    dcrnn_prep_weights<<<6, 256, 0, stream>>>(Wz, Wr, Wh, wbuf);

    const int numTiles = nNodes >> 4;                // full 16-node tiles
    dcrnn_gru_kernel<<<1024, 256, 0, stream>>>(x, h, bz, br, bh, wlin, blin,
                                               wbuf, out_head, out_h, numTiles);

    const int rem = nNodes & 15;                     // 0 for N=500000
    if (rem > 0) {
        dcrnn_tail_kernel<<<1, 16, 0, stream>>>(x, h, Wz, bz, Wr, br, Wh, bh,
                                                wlin, blin, out_head, out_h,
                                                numTiles << 4, nNodes);
    }
}
